// GraphEncoder_20641612824817
// MI455X (gfx1250) — matrix operand order repaired and checked by reading, not compiled, not run on hardware
//
#include <hip/hip_runtime.h>
#include <hip/hip_bf16.h>
#include <math.h>

typedef __attribute__((ext_vector_type(16))) _Float16 v16h;
typedef __attribute__((ext_vector_type(8)))  float    v8f;

#define NEG_SLOPE 0.2f
#define TPB 256

// ---------------------------------------------------------------------------
// Float atomic max via monotonic int mapping (init with -FLT_MAX).
// ---------------------------------------------------------------------------
__device__ __forceinline__ void atomicMaxF(float* addr, float val) {
  int iv = __float_as_int(val);
  if (iv >= 0) {
    atomicMax((int*)addr, iv);
  } else {
    atomicMin((unsigned int*)addr, (unsigned int)iv);
  }
}

__global__ void fill_kernel(float* __restrict__ p, float v, size_t n) {
  size_t i = (size_t)blockIdx.x * blockDim.x + threadIdx.x;
  if (i < n) p[i] = v;
}

// ---------------------------------------------------------------------------
// C[M x Nc] = A[M x K] * B[K x Nc], f32 in/out, f16 WMMA compute (f32 accum).
// K is compile-time -> fully unrolled K loop (8 static v_wmma per wave).
// Each wave computes a 16x32 output supertile: one shared A fragment feeds
// two B fragments / two accumulators (register blocking, halves A traffic).
// Requires M%16==0, Nc%32==0, K%32==0.
// Per-lane layouts (A/B use contiguous K, not the ISA 7.12.2 table order):
//   A (16x32 f16): lane l<16 holds row M=l; per lane two contiguous 8-elem
//     K chunks at K = 16*half and K = 16*half + 8.
//   B (32x16 f16): lanes = N, VGPR v -> K pair (2v,2v+1), lane half adds K+16.
//   C/D: VGPR r -> row (r + 8*half), lanes 0..15 = N.
// ---------------------------------------------------------------------------
template <int K>
__global__ void gemm_wmma_f16(const float* __restrict__ A, const float* __restrict__ B,
                              float* __restrict__ C, int M, int Nc) {
  int lane = threadIdx.x & 31;
  int wib  = threadIdx.x >> 5;
  int ntn  = Nc >> 5;                       // 32-column supertiles
  int total = (M >> 4) * ntn;
  int tile = blockIdx.x * (blockDim.x >> 5) + wib;
  if (tile >= total) return;
  int mt = tile / ntn, nt = tile % ntn;
  int half = lane >> 4, l16 = lane & 15;
  int row  = (mt << 4) + l16;               // A row held by this lane
  int col0 = nt << 5;

  v8f acc0 = {}, acc1 = {};
#pragma unroll
  for (int k0 = 0; k0 < K; k0 += 32) {
    v16h a, b0, b1;
    // A fragment: two contiguous 8-float chunks (vectorizable to b128 loads)
    const float* ap = A + (size_t)row * K + k0 + (half << 4);
#pragma unroll
    for (int j = 0; j < 8; ++j) a[j] = (_Float16)ap[j];        // K = k0+16h+j
#pragma unroll
    for (int j = 0; j < 8; ++j) a[8 + j] = (_Float16)ap[8 + j]; // K = k0+16h+8+j
    // B fragments for columns [col0, col0+16) and [col0+16, col0+32)
#pragma unroll
    for (int v = 0; v < 8; ++v) {
      int kk = k0 + (half << 4) + (v << 1);
      const float* bp = B + (size_t)kk * Nc + col0 + l16;
      b0[2 * v]     = (_Float16)bp[0];
      b0[2 * v + 1] = (_Float16)bp[Nc];
      b1[2 * v]     = (_Float16)bp[16];
      b1[2 * v + 1] = (_Float16)bp[Nc + 16];
    }
    acc0 = __builtin_amdgcn_wmma_f32_16x16x32_f16(false, a, false, b0,
                                                  (short)0, acc0, false, false);
    acc1 = __builtin_amdgcn_wmma_f32_16x16x32_f16(false, a, false, b1,
                                                  (short)0, acc1, false, false);
  }
#pragma unroll
  for (int r = 0; r < 8; ++r) {
    int m = (mt << 4) + r + (half << 3);
    float* cp = C + (size_t)m * Nc + col0 + l16;
    cp[0]  = acc0[r];
    cp[16] = acc1[r];
  }
}

// ---------------------------------------------------------------------------
// Per-node attention scores: a_src[n,h] = sum_c h[n,h,c]*att_src[h,c] (C=32).
// One wave (32 lanes) per node; shuffle reduction per head.
// ---------------------------------------------------------------------------
__global__ void att_scores(const float* __restrict__ h,
                           const float* __restrict__ att_src,
                           const float* __restrict__ att_dst,
                           float* __restrict__ aS, float* __restrict__ aD,
                           int n, int H) {
  int gid = blockIdx.x * blockDim.x + threadIdx.x;
  int node = gid >> 5, lane = gid & 31;
  if (node >= n) return;
  const float* hp = h + (size_t)node * H * 32;
  for (int hh = 0; hh < H; ++hh) {
    float v = hp[hh * 32 + lane];
    float s = v * att_src[hh * 32 + lane];
    float d = v * att_dst[hh * 32 + lane];
#pragma unroll
    for (int o = 16; o > 0; o >>= 1) {
      s += __shfl_down(s, o, 32);
      d += __shfl_down(d, o, 32);
    }
    if (lane == 0) {
      aS[(size_t)node * H + hh] = s;
      aD[(size_t)node * H + hh] = d;
    }
  }
}

// ---------------------------------------------------------------------------
// Edge pass 1: segment max of leaky_relu(a_src[s]+a_dst[d]) over dst.
// ---------------------------------------------------------------------------
__global__ void edge_max(const int* __restrict__ ei, int E, int N,
                         const float* __restrict__ aS, const float* __restrict__ aD,
                         float* __restrict__ mx, int H) {
  int e = blockIdx.x * blockDim.x + threadIdx.x;
  if (e >= E + N) return;
  int s = (e < E) ? ei[e] : (e - E);
  int d = (e < E) ? ei[E + e] : (e - E);
  for (int h = 0; h < H; ++h) {
    float v = aS[(size_t)s * H + h] + aD[(size_t)d * H + h];
    v = v > 0.f ? v : NEG_SLOPE * v;
    atomicMaxF(&mx[(size_t)d * H + h], v);
  }
}

// ---------------------------------------------------------------------------
// Edge pass 2: ex = exp(e - m[dst]); store per-edge, atomic-sum denom per dst.
// ---------------------------------------------------------------------------
__global__ void edge_expsum(const int* __restrict__ ei, int E, int N,
                            const float* __restrict__ aS, const float* __restrict__ aD,
                            const float* __restrict__ mx,
                            float* __restrict__ ex, float* __restrict__ dn, int H) {
  int e = blockIdx.x * blockDim.x + threadIdx.x;
  if (e >= E + N) return;
  int s = (e < E) ? ei[e] : (e - E);
  int d = (e < E) ? ei[E + e] : (e - E);
  for (int h = 0; h < H; ++h) {
    float v = aS[(size_t)s * H + h] + aD[(size_t)d * H + h];
    v = v > 0.f ? v : NEG_SLOPE * v;
    float p = expf(v - mx[(size_t)d * H + h]);
    ex[(size_t)e * H + h] = p;
    atomicAdd(&dn[(size_t)d * H + h], p);
  }
}

// ---------------------------------------------------------------------------
// Edge pass 3: out[dst] += alpha * h[src]. One wave per edge, lane = channel.
// ---------------------------------------------------------------------------
__global__ void edge_aggr(const int* __restrict__ ei, int E, int N,
                          const float* __restrict__ ex, const float* __restrict__ dn,
                          const float* __restrict__ hf, float* __restrict__ outp, int H) {
  long long gid = (long long)blockIdx.x * blockDim.x + threadIdx.x;
  int e = (int)(gid >> 5), lane = (int)(gid & 31);
  if (e >= E + N) return;
  int s = (e < E) ? ei[e] : (e - E);
  int d = (e < E) ? ei[E + e] : (e - E);
  const float* hrow = hf + (size_t)s * H * 32;
  __builtin_prefetch(hrow, 0, 1);   // global_prefetch of the gathered row
  for (int h = 0; h < H; ++h) {
    float alpha = ex[(size_t)e * H + h] / (dn[(size_t)d * H + h] + 1e-16f);
    float val = alpha * hrow[h * 32 + lane];
    atomicAdd(&outp[(size_t)d * H * 32 + h * 32 + lane], val);
  }
}

__global__ void bias_elu(float* __restrict__ h, const float* __restrict__ b,
                         int n, int F) {
  size_t i = (size_t)blockIdx.x * blockDim.x + threadIdx.x;
  if (i >= (size_t)n * F) return;
  float v = h[i] + b[i % F];
  h[i] = v > 0.f ? v : expm1f(v);
}

// ---------------------------------------------------------------------------
// Gate MLP: relu(h @ g_w1 + g_b1) @ g_w2 + g_b2. One wave per node (HID=32).
// ---------------------------------------------------------------------------
__global__ void gate_kernel(const float* __restrict__ h2,
                            const float* __restrict__ gw1, const float* __restrict__ gb1,
                            const float* __restrict__ gw2, const float* __restrict__ gb2,
                            float* __restrict__ gate, int n) {
  int gid = blockIdx.x * blockDim.x + threadIdx.x;
  int node = gid >> 5, lane = gid & 31;
  if (node >= n) return;
  const float* hp = h2 + (size_t)node * 32;
  float t = gb1[lane];
#pragma unroll 8
  for (int i = 0; i < 32; ++i) t += hp[i] * gw1[i * 32 + lane];
  t = fmaxf(t, 0.f);
  float g = t * gw2[lane];
#pragma unroll
  for (int o = 16; o > 0; o >>= 1) g += __shfl_down(g, o, 32);
  if (lane == 0) gate[node] = g + gb2[0];
}

__global__ void pool_max(const float* __restrict__ gate, const int* __restrict__ batch,
                         float* __restrict__ gm, int n) {
  int i = blockIdx.x * blockDim.x + threadIdx.x;
  if (i < n) atomicMaxF(&gm[batch[i]], gate[i]);
}

__global__ void pool_sum(const float* __restrict__ gate, const int* __restrict__ batch,
                         const float* __restrict__ gm, float* __restrict__ gs, int n) {
  int i = blockIdx.x * blockDim.x + threadIdx.x;
  if (i >= n) return;
  int b = batch[i];
  atomicAdd(&gs[b], expf(gate[i] - gm[b]));
}

__global__ void pool_out(const float* __restrict__ gate, const int* __restrict__ batch,
                         const float* __restrict__ gm, const float* __restrict__ gs,
                         const float* __restrict__ h2, float* __restrict__ outp, int n) {
  int gid = blockIdx.x * blockDim.x + threadIdx.x;
  int node = gid >> 5, lane = gid & 31;
  if (node >= n) return;
  int b = batch[node];
  float w = expf(gate[node] - gm[b]) / (gs[b] + 1e-16f);
  atomicAdd(&outp[(size_t)b * 32 + lane], w * h2[(size_t)node * 32 + lane]);
}

// ---------------------------------------------------------------------------
extern "C" void kernel_launch(void* const* d_in, const int* in_sizes, int n_in,
                              void* d_out, int out_size, void* d_ws, size_t ws_size,
                              hipStream_t stream) {
  const float* x        = (const float*)d_in[0];
  const int*   ei       = (const int*)d_in[1];
  const int*   batch    = (const int*)d_in[2];
  const float* W1       = (const float*)d_in[3];
  const float* attS1    = (const float*)d_in[4];
  const float* attD1    = (const float*)d_in[5];
  const float* b1       = (const float*)d_in[6];
  const float* W2       = (const float*)d_in[7];
  const float* attS2    = (const float*)d_in[8];
  const float* attD2    = (const float*)d_in[9];
  const float* b2       = (const float*)d_in[10];
  const float* gw1      = (const float*)d_in[11];
  const float* gb1      = (const float*)d_in[12];
  const float* gw2      = (const float*)d_in[13];
  const float* gb2      = (const float*)d_in[14];
  float* outp = (float*)d_out;

  const int EMB = 128, HID = 32, H1 = 4;
  const int N    = in_sizes[0] / EMB;
  const int E    = in_sizes[1] / 2;
  const int Etot = E + N;
  const float NEGINF = -3.402823466e+38f;

  // Workspace layout (floats)
  float* ws    = (float*)d_ws;
  float* h1raw = ws;                        // N*128
  float* h1act = h1raw + (size_t)N * 128;   // N*128 (agg1 dest, ELU in place)
  float* aS    = h1act + (size_t)N * 128;   // N*4
  float* aD    = aS + (size_t)N * 4;        // N*4
  float* mx    = aD + (size_t)N * 4;        // N*4
  float* dn    = mx + (size_t)N * 4;        // N*4
  float* ex    = dn + (size_t)N * 4;        // Etot*4
  float* gate  = ex + (size_t)Etot * 4;     // N
  float* gm    = gate + N;                  // 64
  float* gs    = gm + 64;                   // 64
  float* h2raw = h1raw;                     // N*32 (reuse: h1raw dead after aggr1)
  float* h2act = h1raw + (size_t)N * 32;    // N*32

  auto blocks = [](long long t) { return (unsigned)((t + TPB - 1) / TPB); };

  // ---- Layer 1 ----
  fill_kernel<<<blocks((long long)N * 128), TPB, 0, stream>>>(h1act, 0.f, (size_t)N * 128);
  fill_kernel<<<blocks((long long)N * H1), TPB, 0, stream>>>(mx, NEGINF, (size_t)N * H1);
  fill_kernel<<<blocks((long long)N * H1), TPB, 0, stream>>>(dn, 0.f, (size_t)N * H1);

  {
    int tiles = (N / 16) * (EMB / 32);            // 3125 * 4 supertiles
    gemm_wmma_f16<128><<<(tiles + 7) / 8, TPB, 0, stream>>>(x, W1, h1raw, N, EMB);
  }
  att_scores<<<blocks((long long)N * 32), TPB, 0, stream>>>(h1raw, attS1, attD1, aS, aD, N, H1);
  edge_max<<<blocks(Etot), TPB, 0, stream>>>(ei, E, N, aS, aD, mx, H1);
  edge_expsum<<<blocks(Etot), TPB, 0, stream>>>(ei, E, N, aS, aD, mx, ex, dn, H1);
  edge_aggr<<<blocks((long long)Etot * 32), TPB, 0, stream>>>(ei, E, N, ex, dn, h1raw, h1act, H1);
  bias_elu<<<blocks((long long)N * 128), TPB, 0, stream>>>(h1act, b1, N, 128);

  // ---- Layer 2 ----
  fill_kernel<<<blocks((long long)N * 32), TPB, 0, stream>>>(h2act, 0.f, (size_t)N * 32);
  fill_kernel<<<blocks(N), TPB, 0, stream>>>(mx, NEGINF, (size_t)N);
  fill_kernel<<<blocks(N), TPB, 0, stream>>>(dn, 0.f, (size_t)N);

  {
    int tiles = (N / 16) * (HID / 32);            // 3125 * 1 supertiles
    gemm_wmma_f16<128><<<(tiles + 7) / 8, TPB, 0, stream>>>(h1act, W2, h2raw, N, HID);
  }
  att_scores<<<blocks((long long)N * 32), TPB, 0, stream>>>(h2raw, attS2, attD2, aS, aD, N, 1);
  edge_max<<<blocks(Etot), TPB, 0, stream>>>(ei, E, N, aS, aD, mx, 1);
  edge_expsum<<<blocks(Etot), TPB, 0, stream>>>(ei, E, N, aS, aD, mx, ex, dn, 1);
  edge_aggr<<<blocks((long long)Etot * 32), TPB, 0, stream>>>(ei, E, N, ex, dn, h2raw, h2act, 1);
  bias_elu<<<blocks((long long)N * 32), TPB, 0, stream>>>(h2act, b2, N, 32);

  // ---- Attentional pooling over graphs ----
  gate_kernel<<<blocks((long long)N * 32), TPB, 0, stream>>>(h2act, gw1, gb1, gw2, gb2, gate, N);
  fill_kernel<<<1, TPB, 0, stream>>>(gm, NEGINF, 64);
  fill_kernel<<<1, TPB, 0, stream>>>(gs, 0.f, 64);
  fill_kernel<<<blocks(out_size), TPB, 0, stream>>>(outp, 0.f, (size_t)out_size);
  pool_max<<<blocks(N), TPB, 0, stream>>>(gate, batch, gm, N);
  pool_sum<<<blocks(N), TPB, 0, stream>>>(gate, batch, gm, gs, N);
  pool_out<<<blocks((long long)N * 32), TPB, 0, stream>>>(gate, batch, gm, gs, h2act, outp, N);
}
